// TransformerBlock_20736102105578
// MI455X (gfx1250) — compile-verified, matrix operand order checked by reading, not run on hardware
//
#include <hip/hip_runtime.h>
#include <hip/hip_bf16.h>

// ---------------------------------------------------------------------------
// TransformerBlock on MI455X (gfx1250): all matmuls via v_wmma_f32_16x16x32_f16,
// f16 tile staging via GLOBAL_LOAD_ASYNC_TO_LDS_B128 (ASYNCcnt DMA path).
// ---------------------------------------------------------------------------
// Shapes: b=2, s=2048 -> M = 4096 tokens, HID=1024, HEADS=16, HEAD_DIM=64,
// INTER=4096.  fp32 in/out, f16 WMMA with f32 accumulation internally.

typedef __attribute__((ext_vector_type(16))) _Float16 v16h;
typedef __attribute__((ext_vector_type(8)))  float    v8f;
typedef __attribute__((ext_vector_type(4)))  _Float16 v4h;

#define TOKENS 4096
#define HID    1024
#define HEADS  16
#define HDIM   64
#define INTER  4096
#define SEQ    2048
#define ATTN_SCALE 0.125f   // 1/sqrt(64)

union FragU { uint4 u[2]; v16h h; };

// Load one 16x32 f16 A/B fragment from LDS (row-major rows, n-major for B).
// Per CDNA5 ISA 16-bit A layout: lanes 0-15 hold K {0..7,16..23},
// lanes 16-31 hold K {8..15,24..31}; two 16B ds_load_b128 per fragment.
__device__ __forceinline__ v16h ld_frag(const _Float16* rowbase, int k0, int lane) {
  const _Float16* p = rowbase + k0 + ((lane >> 4) << 3);
  FragU f;
  f.u[0] = *(const uint4*)(p);
  f.u[1] = *(const uint4*)(p + 16);
  return f.h;
}

__device__ __forceinline__ v8f v8f_zero() {
  v8f z;
#pragma unroll
  for (int i = 0; i < 8; ++i) z[i] = 0.0f;
  return z;
}

#define WMMA_F16(Afrag, Bfrag, Cacc) \
  __builtin_amdgcn_wmma_f32_16x16x32_f16(false, (Afrag), false, (Bfrag), (short)0, (Cacc), false, false)

// CDNA5 async DMA: copy 16 bytes per lane, global -> LDS, no VGPR roundtrip.
// VDST = 32-bit LDS byte address (low 32 bits of flat shared-aperture ptr),
// ADDR = 64-bit global address, SADDR = off.  Tracked by ASYNCcnt.
__device__ __forceinline__ void async_g2l_b128(void* lds_dst, const void* gsrc) {
  asm volatile("global_load_async_to_lds_b128 %0, %1, off"
               :: "v"((unsigned int)(size_t)lds_dst), "v"(gsrc)
               : "memory");
}
__device__ __forceinline__ void async_wait0() {
  asm volatile("s_wait_asynccnt 0x0" ::: "memory");
}

// ---------------------------------------------------------------------------
// RMSNorm + cast to f16.  One block per token row (HID = 1024, 256 threads).
// ---------------------------------------------------------------------------
__global__ __launch_bounds__(256) void rmsnorm_cast_kernel(
    const float* __restrict__ X, const float* __restrict__ W,
    _Float16* __restrict__ Out) {
  __shared__ float red[8];
  const int row  = blockIdx.x;
  const int tid  = threadIdx.x;
  const float* xr = X + (size_t)row * HID;
  float v[4];
  float ss = 0.0f;
#pragma unroll
  for (int j = 0; j < 4; ++j) {
    int i = tid + j * 256;
    v[j] = xr[i];
    ss += v[j] * v[j];
  }
#pragma unroll
  for (int m = 16; m >= 1; m >>= 1) ss += __shfl_xor(ss, m, 32);
  if ((tid & 31) == 0) red[tid >> 5] = ss;
  __syncthreads();
  float tot = 0.0f;
#pragma unroll
  for (int w = 0; w < 8; ++w) tot += red[w];
  const float rinv = rsqrtf(tot * (1.0f / (float)HID) + 1e-6f);
#pragma unroll
  for (int j = 0; j < 4; ++j) {
    int i = tid + j * 256;
    Out[(size_t)row * HID + i] = (_Float16)(v[j] * rinv * W[i]);
  }
}

// ---------------------------------------------------------------------------
// WMMA GEMM:  Out[M,N] = A[M,K](f16) @ Bw[N,K](f32 weights, h@W.T form)
// Block tile 128x128, K-step 32, 256 threads = 8 waves, each wave 64x32.
// A tile staged via async DMA; B tile converted f32->f16 through VGPRs.
// ---------------------------------------------------------------------------
enum { EPI_F16 = 0, EPI_F16_T = 1, EPI_F32_RES = 2, EPI_SILU = 3, EPI_MUL = 4 };

#define GLDT 40   // LDS row pitch in halfs (32 data + 8 pad; 80B = 5*16B aligned)

template <int EPI>
__global__ __launch_bounds__(256) void gemm_h16_kernel(
    const _Float16* __restrict__ A, const float* __restrict__ Bw,
    const float* __restrict__ Res, const _Float16* __restrict__ Mul,
    void* __restrict__ OutP, int M, int N, int K) {
  __shared__ _Float16 As[128 * GLDT];
  __shared__ _Float16 Bs[128 * GLDT];
  const int tid  = threadIdx.x;
  const int lane = tid & 31;
  const int wave = tid >> 5;
  const int bm = blockIdx.y * 128;
  const int bn = blockIdx.x * 128;
  const int wm = (wave & 1) * 64;   // 2 waves along M
  const int wn = (wave >> 1) * 32;  // 4 waves along N

  v8f acc[4][2];
#pragma unroll
  for (int mt = 0; mt < 4; ++mt)
#pragma unroll
    for (int nt = 0; nt < 2; ++nt) acc[mt][nt] = v8f_zero();

  for (int k0 = 0; k0 < K; k0 += 32) {
    // Stage A tile: 128 rows x 32 halfs, async global->LDS DMA (2 x B128/thread)
#pragma unroll
    for (int i = 0; i < 2; ++i) {
      int idx = tid + i * 256;            // 512 x 16B total
      int r = idx >> 2, c = idx & 3;      // 4 x 16B per row
      async_g2l_b128(As + r * GLDT + c * 8,
                     A + (size_t)(bm + r) * K + k0 + c * 8);
    }
    // Stage B tile: 128 rows(n-major) x 32 f32 -> f16 convert on the fly
#pragma unroll
    for (int i = 0; i < 4; ++i) {
      int idx = tid + i * 256;            // 1024 float4 total
      int r = idx >> 3, c = idx & 7;      // 8 float4 per row
      float4 f = *(const float4*)(Bw + (size_t)(bn + r) * K + k0 + c * 4);
      v4h hv;
      hv[0] = (_Float16)f.x; hv[1] = (_Float16)f.y;
      hv[2] = (_Float16)f.z; hv[3] = (_Float16)f.w;
      *(v4h*)(Bs + r * GLDT + c * 4) = hv;
    }
    // Prefetch next weight K-tile (global_prefetch_b8)
    if (k0 + 32 < K) {
      __builtin_prefetch(Bw + (size_t)(bn + (tid >> 1)) * K + k0 + 32, 0, 3);
    }
    async_wait0();
    __syncthreads();

    v16h af[4], bf[2];
#pragma unroll
    for (int mt = 0; mt < 4; ++mt)
      af[mt] = ld_frag(As + (wm + mt * 16 + (lane & 15)) * GLDT, 0, lane);
#pragma unroll
    for (int nt = 0; nt < 2; ++nt)
      bf[nt] = ld_frag(Bs + (wn + nt * 16 + (lane & 15)) * GLDT, 0, lane);
#pragma unroll
    for (int mt = 0; mt < 4; ++mt)
#pragma unroll
      for (int nt = 0; nt < 2; ++nt)
        acc[mt][nt] = WMMA_F16(af[mt], bf[nt], acc[mt][nt]);
    __syncthreads();
  }

  // Epilogue.  C/D layout: VGPR v, lanes 0-15 -> M=v, lanes 16-31 -> M=v+8.
  const int mrow0 = bm + wm + ((lane >> 4) << 3);
  const int ncol0 = bn + wn + (lane & 15);
#pragma unroll
  for (int mt = 0; mt < 4; ++mt) {
#pragma unroll
    for (int nt = 0; nt < 2; ++nt) {
#pragma unroll
      for (int v = 0; v < 8; ++v) {
        const int m = mrow0 + mt * 16 + v;
        const int n = ncol0 + nt * 16;
        const float val = acc[mt][nt][v];
        if constexpr (EPI == EPI_F16) {
          ((_Float16*)OutP)[(size_t)m * N + n] = (_Float16)val;
        } else if constexpr (EPI == EPI_F16_T) {
          ((_Float16*)OutP)[(size_t)n * M + m] = (_Float16)val;  // V^T store
        } else if constexpr (EPI == EPI_F32_RES) {
          ((float*)OutP)[(size_t)m * N + n] = val + Res[(size_t)m * N + n];
        } else if constexpr (EPI == EPI_SILU) {
          const float s = val / (1.0f + __expf(-val));
          ((_Float16*)OutP)[(size_t)m * N + n] = (_Float16)s;
        } else {  // EPI_MUL: gate * up (in-place safe: same element per thread)
          const float g = (float)Mul[(size_t)m * N + n];
          ((_Float16*)OutP)[(size_t)m * N + n] = (_Float16)(val * g);
        }
      }
    }
  }
}

// ---------------------------------------------------------------------------
// Flash attention: one block = (b, head, 64-query tile), 128 threads = 4 waves,
// each wave owns 16 query rows.  Online softmax, WMMA for QK^T and PV.
// Q/K/V tiles staged via async global->LDS DMA.
// ---------------------------------------------------------------------------
#define ALD 72  // 64 halfs + 8 pad; 144B = 9*16B aligned

__global__ __launch_bounds__(128) void attn_kernel(
    const _Float16* __restrict__ Q, const _Float16* __restrict__ Km,
    const _Float16* __restrict__ VT, _Float16* __restrict__ O) {
  __shared__ _Float16 Qs[64 * ALD];
  __shared__ _Float16 Ks[64 * ALD];
  __shared__ _Float16 Vs[64 * ALD];        // V^T tile: [d][kk]
  __shared__ _Float16 Ps[4 * 16 * ALD];    // per-wave P transpose patch
  const int tid = threadIdx.x, lane = tid & 31, wave = tid >> 5;
  const int qb = blockIdx.x, h = blockIdx.y, b = blockIdx.z;
  const size_t tokBase = (size_t)b * SEQ;

  // Load Q tile (64 x 64 halfs) via async DMA
#pragma unroll
  for (int i = 0; i < 4; ++i) {
    int idx = tid + i * 128;               // 512 x 16B
    int r = idx >> 3, c = idx & 7;
    async_g2l_b128(Qs + r * ALD + c * 8,
                   Q + (tokBase + qb * 64 + r) * HID + h * HDIM + c * 8);
  }
  async_wait0();
  __syncthreads();
  v16h qf[2];
  {
    const _Float16* qrow = Qs + (wave * 16 + (lane & 15)) * ALD;
    qf[0] = ld_frag(qrow, 0, lane);
    qf[1] = ld_frag(qrow, 32, lane);
  }

  float mrow[8], lrow[8];
  v8f oacc[4];
#pragma unroll
  for (int r = 0; r < 8; ++r) { mrow[r] = -1e30f; lrow[r] = 0.0f; }
#pragma unroll
  for (int t = 0; t < 4; ++t) oacc[t] = v8f_zero();

  for (int kb = 0; kb < SEQ / 64; ++kb) {
    __syncthreads();  // protect Ks/Vs reuse across iterations
#pragma unroll
    for (int i = 0; i < 4; ++i) {
      int idx = tid + i * 128;
      int r = idx >> 3, c = idx & 7;
      async_g2l_b128(Ks + r * ALD + c * 8,
                     Km + (tokBase + kb * 64 + r) * HID + h * HDIM + c * 8);
      async_g2l_b128(Vs + r * ALD + c * 8,
                     VT + (size_t)(h * HDIM + r) * TOKENS + tokBase + kb * 64 + c * 8);
    }
    async_wait0();
    __syncthreads();

    // S = Q @ K^T  (K tile is n-major as stored: Ks[kk][d])
    v8f sacc[4];
#pragma unroll
    for (int nt = 0; nt < 4; ++nt) sacc[nt] = v8f_zero();
#pragma unroll
    for (int nt = 0; nt < 4; ++nt) {
      const _Float16* krow = Ks + (nt * 16 + (lane & 15)) * ALD;
      v16h kf0 = ld_frag(krow, 0, lane);
      v16h kf1 = ld_frag(krow, 32, lane);
      sacc[nt] = WMMA_F16(qf[0], kf0, sacc[nt]);
      sacc[nt] = WMMA_F16(qf[1], kf1, sacc[nt]);
    }

    // Online softmax per query row.  Row r of this half-wave lives in VGPR r
    // across 16 lanes (4 column tiles).  Reductions via shfl_xor within half.
    _Float16* pbase = Ps + wave * 16 * ALD;
#pragma unroll
    for (int r = 0; r < 8; ++r) {
      float sv[4];
#pragma unroll
      for (int nt = 0; nt < 4; ++nt) sv[nt] = sacc[nt][r] * ATTN_SCALE;
      float tm = fmaxf(fmaxf(sv[0], sv[1]), fmaxf(sv[2], sv[3]));
#pragma unroll
      for (int m = 8; m >= 1; m >>= 1) tm = fmaxf(tm, __shfl_xor(tm, m, 32));
      const float mnew = fmaxf(mrow[r], tm);
      const float corr = __expf(mrow[r] - mnew);
      mrow[r] = mnew;
      float ps = 0.0f;
      const int prow = r + ((lane >> 4) << 3);
#pragma unroll
      for (int nt = 0; nt < 4; ++nt) {
        const float p = __expf(sv[nt] - mnew);
        ps += p;
        pbase[prow * ALD + nt * 16 + (lane & 15)] = (_Float16)p;  // transpose via LDS
      }
#pragma unroll
      for (int m = 8; m >= 1; m >>= 1) ps += __shfl_xor(ps, m, 32);
      lrow[r] = lrow[r] * corr + ps;
#pragma unroll
      for (int t = 0; t < 4; ++t) oacc[t][r] *= corr;
    }

    // O += P @ V  (Vs is already n-major: Vs[d][kk])
    {
      const _Float16* prow2 = pbase + (lane & 15) * ALD;
      v16h pf0 = ld_frag(prow2, 0, lane);
      v16h pf1 = ld_frag(prow2, 32, lane);
#pragma unroll
      for (int nt = 0; nt < 4; ++nt) {
        const _Float16* vrow = Vs + (nt * 16 + (lane & 15)) * ALD;
        v16h vf0 = ld_frag(vrow, 0, lane);
        v16h vf1 = ld_frag(vrow, 32, lane);
        oacc[nt] = WMMA_F16(pf0, vf0, oacc[nt]);
        oacc[nt] = WMMA_F16(pf1, vf1, oacc[nt]);
      }
    }
  }

  // Normalize and store (token-major f16, per-head column slice)
#pragma unroll
  for (int r = 0; r < 8; ++r) {
    const float inv = 1.0f / lrow[r];
    const int m = wave * 16 + r + ((lane >> 4) << 3);
    const size_t base = (tokBase + qb * 64 + m) * HID + h * HDIM + (lane & 15);
#pragma unroll
    for (int nt = 0; nt < 4; ++nt)
      O[base + nt * 16] = (_Float16)(oacc[nt][r] * inv);
  }
}

// ---------------------------------------------------------------------------
// Host-side orchestration
// ---------------------------------------------------------------------------
extern "C" void kernel_launch(void* const* d_in, const int* in_sizes, int n_in,
                              void* d_out, int out_size, void* d_ws, size_t ws_size,
                              hipStream_t stream) {
  const float* x       = (const float*)d_in[0];
  const float* wq      = (const float*)d_in[1];
  const float* wk      = (const float*)d_in[2];
  const float* wv      = (const float*)d_in[3];
  const float* wo      = (const float*)d_in[4];
  const float* w_gate  = (const float*)d_in[5];
  const float* w_up    = (const float*)d_in[6];
  const float* w_down  = (const float*)d_in[7];
  const float* norm1_w = (const float*)d_in[8];
  const float* norm2_w = (const float*)d_in[9];

  char* ws = (char*)d_ws;
  size_t off = 0;
  auto alloc = [&](size_t bytes) -> void* {
    void* p = ws + off;
    off += (bytes + 255) & ~(size_t)255;
    return p;
  };
  _Float16* h16    = (_Float16*)alloc((size_t)TOKENS * HID * 2);
  _Float16* q16    = (_Float16*)alloc((size_t)TOKENS * HID * 2);
  _Float16* k16    = (_Float16*)alloc((size_t)TOKENS * HID * 2);
  _Float16* vT16   = (_Float16*)alloc((size_t)TOKENS * HID * 2);  // [hid][tokens]
  _Float16* attn16 = (_Float16*)alloc((size_t)TOKENS * HID * 2);
  float*    x1f    = (float*)   alloc((size_t)TOKENS * HID * 4);
  _Float16* h2_16  = (_Float16*)alloc((size_t)TOKENS * HID * 2);
  _Float16* g16    = (_Float16*)alloc((size_t)TOKENS * INTER * 2); // gate, then gate*up

  const dim3 blk256(256), blk128(128);
  const dim3 gHID(HID / 128, TOKENS / 128);    // N=1024
  const dim3 gINT(INTER / 128, TOKENS / 128);  // N=4096
  const dim3 gAttn(SEQ / 64, HEADS, 2);

  // 1) h = rmsnorm(x) * norm1_w  (f16)
  rmsnorm_cast_kernel<<<TOKENS, blk256, 0, stream>>>(x, norm1_w, h16);
  // 2-4) q/k/v projections (V stored transposed for the PV WMMA operand)
  gemm_h16_kernel<EPI_F16><<<gHID, blk256, 0, stream>>>(h16, wq, nullptr, nullptr, q16, TOKENS, HID, HID);
  gemm_h16_kernel<EPI_F16><<<gHID, blk256, 0, stream>>>(h16, wk, nullptr, nullptr, k16, TOKENS, HID, HID);
  gemm_h16_kernel<EPI_F16_T><<<gHID, blk256, 0, stream>>>(h16, wv, nullptr, nullptr, vT16, TOKENS, HID, HID);
  // 5) attention (flash, online softmax)
  attn_kernel<<<gAttn, blk128, 0, stream>>>(q16, k16, vT16, attn16);
  // 6) x1 = x + attn @ wo^T  (f32)
  gemm_h16_kernel<EPI_F32_RES><<<gHID, blk256, 0, stream>>>(attn16, wo, x, nullptr, x1f, TOKENS, HID, HID);
  // 7) h2 = rmsnorm(x1) * norm2_w  (f16)
  rmsnorm_cast_kernel<<<TOKENS, blk256, 0, stream>>>(x1f, norm2_w, h2_16);
  // 8) gate = silu(h2 @ w_gate^T)  (f16)
  gemm_h16_kernel<EPI_SILU><<<gINT, blk256, 0, stream>>>(h2_16, w_gate, nullptr, nullptr, g16, TOKENS, INTER, HID);
  // 9) gu = gate * (h2 @ w_up^T)  (f16, in-place over gate buffer)
  gemm_h16_kernel<EPI_MUL><<<gINT, blk256, 0, stream>>>(h2_16, w_up, nullptr, g16, g16, TOKENS, INTER, HID);
  // 10) out = x1 + gu @ w_down^T  (f32)
  gemm_h16_kernel<EPI_F32_RES><<<gHID, blk256, 0, stream>>>(g16, w_down, x1f, nullptr, (float*)d_out, TOKENS, HID, INTER);
}
